// NUSpectralConv1D_36910948942375
// MI455X (gfx1250) — matrix operand order repaired and checked by reading, not compiled, not run on hardware
//
#include <hip/hip_runtime.h>
#include <hip/hip_bf16.h>
#include <math.h>
#include <stdint.h>

// ---------------------------------------------------------------------------
// NUSpectralConv1D for MI455X (gfx1250, wave32, WMMA 16x16x32 f16 / f32-acc,
// TDM tensor_load_to_lds for spectral-tile staging)
// ---------------------------------------------------------------------------
// B=16, C=64, N=4096, M=257 (padded to MP=288), K=128, TEMB=256, H=128

typedef __attribute__((ext_vector_type(16))) _Float16 v16h;
typedef __attribute__((ext_vector_type(8)))  _Float16 v8h;
typedef __attribute__((ext_vector_type(4)))  _Float16 v4h;
typedef __attribute__((ext_vector_type(2)))  _Float16 v2h;
typedef __attribute__((ext_vector_type(8)))  float    v8f;
typedef __attribute__((ext_vector_type(4)))  unsigned int u32x4;
typedef __attribute__((ext_vector_type(8)))  int      i32x8;
typedef __attribute__((ext_vector_type(4)))  int      i32x4;

#define BB   16
#define CC   64
#define NN   4096
#define MM   257
#define MP   288
#define KK   128
#define TEMB 256
#define HH   128
#define PI_F 3.14159265358979323846f
#define LPITCH 40   // LDS row pitch in halves (80 bytes, 16B aligned)

// workspace layout (float indices)
#define OFF_KAPPA 0
#define OFF_GC    (OFF_KAPPA + MP)
#define OFF_INVOM (OFF_GC + MP)
#define OFF_GATE  (OFF_INVOM + MP)              // B*MP
#define OFF_TCOND (OFF_GATE + BB*MP)            // B*C
#define OFF_XR    (OFF_TCOND + BB*CC)           // B*C*MP
#define OFF_XI    (OFF_XR + BB*CC*MP)           // B*C*MP
#define OFF_YH    (OFF_XI + BB*CC*MP)           // halves start here

__device__ __forceinline__ float sigmoidf_(float x) { return 1.0f / (1.0f + __expf(-x)); }
__device__ __forceinline__ float softplusf_(float x) {
  return fmaxf(x, 0.0f) + log1pf(__expf(-fabsf(x)));
}
__device__ __forceinline__ float siluf_(float x) { return x * sigmoidf_(x); }

__device__ __forceinline__ v16h cat8(v8h lo, v8h hi) {
  return __builtin_shufflevector(lo, hi, 0, 1, 2, 3, 4, 5, 6, 7,
                                 8, 9, 10, 11, 12, 13, 14, 15);
}

// Issue a 2-D TDM load: tile_d0 halves x tile_d1 rows, global row stride
// row_stride halves, into LDS at lds_addr with 16B pad after every 64B
// (-> 80B LDS pitch). data_size = 2 bytes.
__device__ __forceinline__ void tdm_load_2d(uint32_t lds_addr, uint64_t gaddr,
                                            uint32_t tile_d0, uint32_t tile_d1,
                                            uint32_t tensor_d0, uint32_t tensor_d1,
                                            uint32_t row_stride) {
  u32x4 g0;
  g0[0] = 1u;                                    // count=1, user descriptor
  g0[1] = lds_addr;                              // lds_addr [63:32]
  g0[2] = (uint32_t)(gaddr & 0xFFFFFFFFull);     // global_addr [95:64]
  g0[3] = ((uint32_t)(gaddr >> 32) & 0x01FFFFFFu) | (2u << 30);  // addr hi | type=2
  i32x8 g1;
  // data_size=1(2B)<<16 | pad_enable<<20 | pad_interval(16 dw)=3<<22 | pad_amount(4 dw)=3<<25
  g1[0] = (int)((1u << 16) | (1u << 20) | (3u << 22) | (3u << 25));
  g1[1] = (int)((tensor_d0 & 0xFFFFu) << 16);            // tensor_dim0 lo16
  g1[2] = (int)(((tensor_d0 >> 16) & 0xFFFFu) | ((tensor_d1 & 0xFFFFu) << 16));
  g1[3] = (int)(((tensor_d1 >> 16) & 0xFFFFu) | ((tile_d0 & 0xFFFFu) << 16));
  g1[4] = (int)(tile_d1 & 0xFFFFu);                      // tile_dim1 | tile_dim2=0
  g1[5] = (int)row_stride;                               // tensor_dim0_stride lo32
  g1[6] = 0;
  g1[7] = 0;
  i32x4 gz = {0, 0, 0, 0};
#if __clang_major__ >= 23
  i32x8 gz8 = {0, 0, 0, 0, 0, 0, 0, 0};
  __builtin_amdgcn_tensor_load_to_lds(g0, g1, gz, gz, gz8, 0);
#else
  __builtin_amdgcn_tensor_load_to_lds(g0, g1, gz, gz, 0);
#endif
}

// ---------------------------------------------------------------------------
// Kernel 1: scalar / small-GEMM setup (one block)
// ---------------------------------------------------------------------------
__global__ __launch_bounds__(256) void k_setup(
    const float* __restrict__ kappa_pos_raw, const float* __restrict__ gabor_c,
    const float* __restrict__ gabor_omega_raw, const float* __restrict__ temb,
    const float* __restrict__ tproj_w, const float* __restrict__ tproj_b,
    const float* __restrict__ mlp_w1, const float* __restrict__ mlp_b1,
    const float* __restrict__ mlp_w2, const float* __restrict__ mlp_b2,
    const float* __restrict__ tau0, const float* __restrict__ tau1_raw,
    const float* __restrict__ alpha_raw, const float* __restrict__ glow_raw,
    const float* __restrict__ gmid_raw, const float* __restrict__ ghigh_raw,
    float* __restrict__ ws) {
  __shared__ float kposS[KK];
  __shared__ float tauS[BB];
  __shared__ float hidS[BB * HH];
  const int t = threadIdx.x;

  if (t == 0) {
    float acc = 0.0f;
    for (int i = 0; i < KK; ++i) { acc += softplusf_(kappa_pos_raw[i]); kposS[i] = acc; }
  }
  __syncthreads();

  // hid = silu(temb @ mlp_w1^T + b1)
  for (int id = t; id < BB * HH; id += 256) {
    const int b = id >> 7, h = id & (HH - 1);
    const float* te = temb + b * TEMB;
    const float* w  = mlp_w1 + h * TEMB;
    float s = mlp_b1[h];
    for (int j = 0; j < TEMB; ++j) s += te[j] * w[j];
    hidS[id] = siluf_(s);
  }
  __syncthreads();
  if (t < BB) {
    float s = mlp_b2[0];
    for (int j = 0; j < HH; ++j) s += hidS[t * HH + j] * mlp_w2[j];
    tauS[t] = tau0[0] + softplusf_(tau1_raw[0]) * tanhf(s);
  }
  __syncthreads();

  // kappa / gabor center / 1/omega (zero-padded to MP)
  for (int m = t; m < MP; m += 256) {
    float kap = 0.0f;
    if (m < KK)            kap = -kposS[KK - 1 - m];
    else if (m == KK)      kap = 0.0f;
    else if (m < MM)       kap = kposS[m - KK - 1];
    ws[OFF_KAPPA + m] = kap;
    ws[OFF_GC + m]    = (m < MM) ? gabor_c[m] : 0.0f;
    ws[OFF_INVOM + m] = (m < MM) ? 1.0f / fmaxf(softplusf_(gabor_omega_raw[m]), 1e-6f) : 0.0f;
  }

  // tcond[b][c] = silu(temb[b]) . tproj_w[c] + tproj_b[c]
  for (int id = t; id < BB * CC; id += 256) {
    const int b = id >> 6, c = id & (CC - 1);
    const float* te = temb + b * TEMB;
    const float* w  = tproj_w + c * TEMB;
    float s = tproj_b[c];
    for (int j = 0; j < TEMB; ++j) s += siluf_(te[j]) * w[j];
    ws[OFF_TCOND + id] = s;
  }

  // gate[b][m]; |kappa| quantiles in closed form (sorted by construction)
  const float th1 = kposS[50]  + 0.4f * (kposS[51]  - kposS[50]);
  const float th2 = kposS[101] + 0.8f * (kposS[102] - kposS[101]);
  const float gl = sigmoidf_(glow_raw[0]);
  const float gm = sigmoidf_(gmid_raw[0]);
  const float gh = sigmoidf_(ghigh_raw[0]);
  const float alpha = softplusf_(alpha_raw[0]) + 1e-12f;
  for (int id = t; id < BB * MP; id += 256) {
    const int b = id / MP, m = id - b * MP;
    float g = 0.0f;
    if (m < MM) {
      float ka;
      if (m < KK)       ka = kposS[KK - 1 - m];
      else if (m == KK) ka = 0.0f;
      else              ka = kposS[m - KK - 1];
      const float band = (ka <= th1) ? gl : ((ka <= th2) ? gm : gh);
      g = band * sigmoidf_(alpha * (tauS[b] - ka));
    }
    ws[OFF_GATE + id] = g;
  }
}

// ---------------------------------------------------------------------------
// Kernel 2: analysis  Xr/Xi[b][c][m] = +/- (1/N) sum_n x~[c][n] * {cw,sw}[n][m]
// grid = (18 m-tiles, 16 b), 8 waves; wave w: ct = w>>1, sin = w&1
// ---------------------------------------------------------------------------
__global__ __launch_bounds__(256) void k_analysis(
    const float* __restrict__ xfeat, const float* __restrict__ z,
    const float* __restrict__ ws, float* __restrict__ Xr, float* __restrict__ Xi) {
  const int mt = blockIdx.x;
  const int b  = blockIdx.y;
  const int t = threadIdx.x;
  const int lane = t & 31;
  const int wv = t >> 5;
  const int ct = wv >> 1;
  const int isSin = wv & 1;

  __shared__ __align__(16) _Float16 xs[CC][LPITCH];      // x(+tcond): [c][n_local]
  __shared__ __align__(16) _Float16 bsa[2][16][LPITCH];  // basis^T: [cw|sw][m_local][n_local]

  // basis thread map: m_local = t>>4 (0..15), n_local pair = (t&15)*2, +1
  const int ml  = t >> 4;
  const int nl0 = (t & 15) * 2;
  const int m   = mt * 16 + ml;
  const float kap = ws[OFF_KAPPA + m];
  const float gcv = ws[OFF_GC + m];
  const float inv = ws[OFF_INVOM + m];

  // x staging map: c = t>>2 (0..63), 8 floats starting at (t&3)*8
  const int xc = t >> 2;
  const int xn = (t & 3) * 8;
  const float tc = ws[OFF_TCOND + b * CC + xc];
  const float* xrow = xfeat + ((size_t)b * CC + xc) * NN + xn;
  const float* zrow = z + (size_t)b * NN;

  v8f acc = {0.f, 0.f, 0.f, 0.f, 0.f, 0.f, 0.f, 0.f};

  for (int n0 = 0; n0 < NN; n0 += 32) {
    __syncthreads();
    // stage x tile (64 x 32) as f16, time conditioning folded in
    {
      const float4 p0 = *(const float4*)(xrow + n0);
      const float4 p1 = *(const float4*)(xrow + n0 + 4);
      v8h hx;
      hx[0] = (_Float16)(p0.x + tc); hx[1] = (_Float16)(p0.y + tc);
      hx[2] = (_Float16)(p0.z + tc); hx[3] = (_Float16)(p0.w + tc);
      hx[4] = (_Float16)(p1.x + tc); hx[5] = (_Float16)(p1.y + tc);
      hx[6] = (_Float16)(p1.z + tc); hx[7] = (_Float16)(p1.w + tc);
      *(v8h*)&xs[xc][xn] = hx;
      if (n0 + 32 < NN) __builtin_prefetch(xrow + n0 + 32, 0, 1);
    }
    // basis tile (32n x 16m), stored transposed [m][n]: 2 positions per thread
    {
      const float zv0 = zrow[n0 + nl0];
      const float zv1 = zrow[n0 + nl0 + 1];
      const float u0 = (zv0 * (1.0f / PI_F) - gcv) * inv;
      const float u1 = (zv1 * (1.0f / PI_F) - gcv) * inv;
      const float w0 = __expf(-0.5f * u0 * u0);
      const float w1 = __expf(-0.5f * u1 * u1);
      const float t0 = zv0 * kap, t1 = zv1 * kap;
      v2h cpair, spair;
      cpair[0] = (_Float16)(__cosf(t0) * w0); cpair[1] = (_Float16)(__cosf(t1) * w1);
      spair[0] = (_Float16)(__sinf(t0) * w0); spair[1] = (_Float16)(__sinf(t1) * w1);
      *(v2h*)&bsa[0][ml][nl0] = cpair;
      *(v2h*)&bsa[1][ml][nl0] = spair;
    }
    __syncthreads();

    // A fragment: two contiguous 16B runs per lane
    const int row  = ct * 16 + (lane & 15);
    const int khiA = ((lane >> 4) & 1) * 8;
    const v16h a = cat8(*(const v8h*)&xs[row][khiA],
                        *(const v8h*)&xs[row][16 + khiA]);
    // B fragment: basis^T row = output column m, K contiguous
    const int col = lane & 15;
    const int kb0 = ((lane >> 4) & 1) * 8;
    const v16h bf = cat8(*(const v8h*)&bsa[isSin][col][kb0],
                         *(const v8h*)&bsa[isSin][col][16 + kb0]);

    acc = __builtin_amdgcn_wmma_f32_16x16x32_f16(false, a, false, bf,
                                                 (short)0, acc, false, false);
  }

  const float sgn = isSin ? (-1.0f / (float)NN) : (1.0f / (float)NN);
  float* dst = isSin ? Xi : Xr;
  const int rbase = ct * 16 + ((lane >> 4) & 1) * 8;
  const int colm  = mt * 16 + (lane & 15);
#pragma unroll
  for (int r = 0; r < 8; ++r) {
    dst[((size_t)b * CC + rbase + r) * MP + colm] = acc[r] * sgn;
  }
}

// ---------------------------------------------------------------------------
// Kernel 3: gate + complex spectral multiply -> Yr, Ym = -Yi (f16)
// ---------------------------------------------------------------------------
__global__ __launch_bounds__(256) void k_cmul(
    const float* __restrict__ Xr, const float* __restrict__ Xi,
    const float* __restrict__ ws, const float* __restrict__ wr,
    const float* __restrict__ wi, _Float16* __restrict__ Yr,
    _Float16* __restrict__ Ym) {
  const int id = blockIdx.x * 256 + threadIdx.x;
  const int total = BB * CC * MP;
  if (id >= total) return;
  const int b = id / (CC * MP);
  const int rem = id - b * CC * MP;
  const int c = rem / MP;
  const int m = rem - c * MP;
  const float g = ws[OFF_GATE + b * MP + m];
  const float xr = Xr[id] * g;
  const float xi = Xi[id] * g;
  float wrv = 0.0f, wiv = 0.0f;
  if (m < MM) { wrv = wr[c * MM + m]; wiv = wi[c * MM + m]; }
  Yr[id] = (_Float16)(xr * wrv - xi * wiv);
  Ym[id] = (_Float16)(-(xr * wiv + xi * wrv));
}

// ---------------------------------------------------------------------------
// Kernel 4: synthesis  out[b][c][n] = sum_m Yr*cw + Ym*sw + bias[c]
// grid = (128 n-chunks of 32, 16 b); wave w: ct = w&3, nt = w>>2
// Y tiles staged by the Tensor Data Mover (pad 16B per 64B -> 80B LDS pitch)
// ---------------------------------------------------------------------------
__global__ __launch_bounds__(256) void k_synthesis(
    const _Float16* __restrict__ Yr, const _Float16* __restrict__ Ym,
    const float* __restrict__ z, const float* __restrict__ ws,
    const float* __restrict__ bias, float* __restrict__ out) {
  const int nchunk = blockIdx.x;
  const int b = blockIdx.y;
  const int t = threadIdx.x;
  const int lane = t & 31;
  const int wv = t >> 5;
  const int ct = wv & 3;
  const int nt = wv >> 2;

  __shared__ __align__(16) _Float16 ys[2][CC][LPITCH];    // TDM-staged Yr / Ym
  __shared__ __align__(16) _Float16 bsn[2][32][LPITCH];   // basis: [cw|sw][n_local][m_local]
  __shared__ float kapS[MP], gcS[MP], ivS[MP];

  // cache basis params in LDS
  for (int mg = t; mg < MP; mg += 256) {
    kapS[mg] = ws[OFF_KAPPA + mg];
    gcS[mg]  = ws[OFF_GC + mg];
    ivS[mg]  = ws[OFF_INVOM + mg];
  }

  // basis thread map: n_local = t>>3 (0..31), 4 consecutive m at (t&7)*4
  const int nlb = t >> 3;
  const int mlb = (t & 7) * 4;
  const float zv = z[(size_t)b * NN + nchunk * 32 + nlb];
  const float zp = zv * (1.0f / PI_F);

  const uint32_t lds_y0 = (uint32_t)(uintptr_t)&ys[0][0][0];
  const uint32_t lds_y1 = (uint32_t)(uintptr_t)&ys[1][0][0];

  v8f acc = {0.f, 0.f, 0.f, 0.f, 0.f, 0.f, 0.f, 0.f};

  for (int kkc = 0; kkc < MP / 32; ++kkc) {
    const int m0 = kkc * 32;
    __syncthreads();
    // wave 0 drives the TDM: two 32x64 f16 tiles, global row stride MP halves
    if (wv == 0) {
      const uint64_t ga0 = (uint64_t)(uintptr_t)(Yr + ((size_t)b * CC) * MP + m0);
      const uint64_t ga1 = (uint64_t)(uintptr_t)(Ym + ((size_t)b * CC) * MP + m0);
      tdm_load_2d(lds_y0, ga0, /*tile*/32, CC, /*tensor*/MP, CC, MP);
      tdm_load_2d(lds_y1, ga1, 32, CC, MP, CC, MP);
    }
    // basis tile (32 n x 32 m), [n][m] layout, 4 consecutive m per thread
    {
      v4h cc, ssv;
#pragma unroll
      for (int j = 0; j < 4; ++j) {
        const int mg = m0 + mlb + j;
        const float u = (zp - gcS[mg]) * ivS[mg];
        const float w = __expf(-0.5f * u * u);
        const float th = zv * kapS[mg];
        cc[j]  = (_Float16)(__cosf(th) * w);
        ssv[j] = (_Float16)(__sinf(th) * w);
      }
      *(v4h*)&bsn[0][nlb][mlb] = cc;
      *(v4h*)&bsn[1][nlb][mlb] = ssv;
    }
    if (wv == 0) __builtin_amdgcn_s_wait_tensorcnt(0);
    __syncthreads();

    // A fragments (Yr / -Yi): rows = c, K contiguous in LDS
    const int row  = ct * 16 + (lane & 15);
    const int khiA = ((lane >> 4) & 1) * 8;
    const v16h ar = cat8(*(const v8h*)&ys[0][row][khiA],
                         *(const v8h*)&ys[0][row][16 + khiA]);
    const v16h am = cat8(*(const v8h*)&ys[1][row][khiA],
                         *(const v8h*)&ys[1][row][16 + khiA]);
    // B fragments: basis rows = output column n, K (=m) contiguous
    const int col = nt * 16 + (lane & 15);
    const int kb0 = ((lane >> 4) & 1) * 8;
    const v16h bc = cat8(*(const v8h*)&bsn[0][col][kb0],
                         *(const v8h*)&bsn[0][col][16 + kb0]);
    const v16h bs = cat8(*(const v8h*)&bsn[1][col][kb0],
                         *(const v8h*)&bsn[1][col][16 + kb0]);

    acc = __builtin_amdgcn_wmma_f32_16x16x32_f16(false, ar, false, bc,
                                                 (short)0, acc, false, false);
    acc = __builtin_amdgcn_wmma_f32_16x16x32_f16(false, am, false, bs,
                                                 (short)0, acc, false, false);
  }

  const int rbase = ct * 16 + ((lane >> 4) & 1) * 8;
  const int ncol  = nchunk * 32 + nt * 16 + (lane & 15);
#pragma unroll
  for (int r = 0; r < 8; ++r) {
    const int c = rbase + r;
    out[((size_t)b * CC + c) * NN + ncol] = acc[r] + bias[c];
  }
}

// ---------------------------------------------------------------------------
extern "C" void kernel_launch(void* const* d_in, const int* in_sizes, int n_in,
                              void* d_out, int out_size, void* d_ws, size_t ws_size,
                              hipStream_t stream) {
  const float* x_feat          = (const float*)d_in[0];
  const float* z               = (const float*)d_in[1];
  const float* temb            = (const float*)d_in[2];
  const float* kappa_pos_raw   = (const float*)d_in[3];
  const float* gabor_c         = (const float*)d_in[4];
  const float* gabor_omega_raw = (const float*)d_in[5];
  const float* wr              = (const float*)d_in[6];
  const float* wi              = (const float*)d_in[7];
  const float* bias            = (const float*)d_in[8];
  const float* tproj_w         = (const float*)d_in[9];
  const float* tproj_b         = (const float*)d_in[10];
  const float* mlp_w1          = (const float*)d_in[11];
  const float* mlp_b1          = (const float*)d_in[12];
  const float* mlp_w2          = (const float*)d_in[13];
  const float* mlp_b2          = (const float*)d_in[14];
  const float* tau0            = (const float*)d_in[15];
  const float* tau1_raw        = (const float*)d_in[16];
  const float* alpha_raw       = (const float*)d_in[17];
  const float* glow            = (const float*)d_in[18];
  const float* gmid            = (const float*)d_in[19];
  const float* ghigh           = (const float*)d_in[20];

  float* ws = (float*)d_ws;
  float* Xr = ws + OFF_XR;
  float* Xi = ws + OFF_XI;
  _Float16* Yr = (_Float16*)(ws + OFF_YH);
  _Float16* Ym = Yr + (size_t)BB * CC * MP;
  float* out = (float*)d_out;

  k_setup<<<1, 256, 0, stream>>>(kappa_pos_raw, gabor_c, gabor_omega_raw, temb,
                                 tproj_w, tproj_b, mlp_w1, mlp_b1, mlp_w2, mlp_b2,
                                 tau0, tau1_raw, alpha_raw, glow, gmid, ghigh, ws);

  k_analysis<<<dim3(MP / 16, BB), 256, 0, stream>>>(x_feat, z, ws, Xr, Xi);

  k_cmul<<<(BB * CC * MP) / 256, 256, 0, stream>>>(Xr, Xi, ws, wr, wi, Yr, Ym);

  k_synthesis<<<dim3(NN / 32, BB), 256, 0, stream>>>(Yr, Ym, z, ws, bias, out);
}
